// CapsuleLayer_72224170049868
// MI455X (gfx1250) — matrix operand order repaired and checked by reading, not compiled, not run on hardware
//
#include <hip/hip_runtime.h>

typedef __attribute__((ext_vector_type(16))) __bf16 v16bf;
typedef __attribute__((ext_vector_type(8)))  float  v8f;

#define BATCH     32
#define IN_CAPS   2048
#define NUM_CAPS  32
#define DIM       32
#define ROUTINGS  3
#define COLS      (NUM_CAPS * DIM)   // 1024 flattened (j,d) columns
#define STRIDE    (COLS + 4)         // pad 4 floats: rows r and r+8 hit different banks

extern "C" __global__ __launch_bounds__(1024)
void capsule_fused_kernel(const float* __restrict__ x,
                          const float* __restrict__ W,
                          float* __restrict__ out)
{
    extern __shared__ float smem[];
    float* u_lds = smem;                       // u_hat[b][c] f32: 32 x 1028
    float* c_lds = smem + BATCH * STRIDE;      // c[b][n]:   32 x 32
    float* o_lds = c_lds + BATCH * NUM_CAPS;   // out[b][d]: 32 x 32

    const int i    = blockIdx.x;               // input capsule
    const int tid  = threadIdx.x;
    const int w    = tid >> 5;                 // wave id 0..31
    const int lane = tid & 31;
    const int half = lane >> 4;
    const int l    = lane & 15;

    // ---- A fragments: x[:, i, :] as two 16x32 bf16 tiles (M=batch, K=32) ----
    // A layout: lane -> M=l; K runs [koffA, koffA+8) and [koffA+16, koffA+24) are
    // linear in the fragment -> two 32B contiguous loads per run pair.
    v16bf afrag[2];
    const int koffA = half * 8;
    #pragma unroll
    for (int mt = 0; mt < 2; ++mt) {
        const float* xp = x + ((size_t)(mt * 16 + l) * IN_CAPS + i) * DIM + koffA;
        const float4 a0 = *reinterpret_cast<const float4*>(xp);
        const float4 a1 = *reinterpret_cast<const float4*>(xp + 4);
        const float4 a2 = *reinterpret_cast<const float4*>(xp + 16);
        const float4 a3 = *reinterpret_cast<const float4*>(xp + 20);
        afrag[mt][0]  = (__bf16)a0.x; afrag[mt][1]  = (__bf16)a0.y;
        afrag[mt][2]  = (__bf16)a0.z; afrag[mt][3]  = (__bf16)a0.w;
        afrag[mt][4]  = (__bf16)a1.x; afrag[mt][5]  = (__bf16)a1.y;
        afrag[mt][6]  = (__bf16)a1.z; afrag[mt][7]  = (__bf16)a1.w;
        afrag[mt][8]  = (__bf16)a2.x; afrag[mt][9]  = (__bf16)a2.y;
        afrag[mt][10] = (__bf16)a2.z; afrag[mt][11] = (__bf16)a2.w;
        afrag[mt][12] = (__bf16)a3.x; afrag[mt][13] = (__bf16)a3.y;
        afrag[mt][14] = (__bf16)a3.z; afrag[mt][15] = (__bf16)a3.w;
    }

    // ---- GEMM: u_hat[b, c] = sum_k x[b,i,k] * W[i, c, k],  c = j*32 + d ----
    const float* Wp = W + (size_t)i * (NUM_CAPS * DIM * DIM);
    const int koffB = half * 8;    // B layout: K runs [koffB, koffB+8) and [koffB+16, koffB+24)
    #pragma unroll
    for (int rep = 0; rep < 2; ++rep) {
        const int nt  = w + rep * 32;
        const int col = nt * 16 + l;
        const float* wp = Wp + (size_t)col * DIM + koffB;
        // unconditional prefetch of the next tile row (wraps inside W[i]; harmless)
        __builtin_prefetch(Wp + (size_t)((col + 512) & (COLS - 1)) * DIM + koffB, 0, 1);

        // two runs of 8 K floats per lane -> 4 x b128 loads + pk cvt
        const float4 b0 = *reinterpret_cast<const float4*>(wp);
        const float4 b1 = *reinterpret_cast<const float4*>(wp + 4);
        const float4 b2 = *reinterpret_cast<const float4*>(wp + 16);
        const float4 b3 = *reinterpret_cast<const float4*>(wp + 20);
        v16bf bfrag;
        bfrag[0]  = (__bf16)b0.x; bfrag[1]  = (__bf16)b0.y;
        bfrag[2]  = (__bf16)b0.z; bfrag[3]  = (__bf16)b0.w;
        bfrag[4]  = (__bf16)b1.x; bfrag[5]  = (__bf16)b1.y;
        bfrag[6]  = (__bf16)b1.z; bfrag[7]  = (__bf16)b1.w;
        bfrag[8]  = (__bf16)b2.x; bfrag[9]  = (__bf16)b2.y;
        bfrag[10] = (__bf16)b2.z; bfrag[11] = (__bf16)b2.w;
        bfrag[12] = (__bf16)b3.x; bfrag[13] = (__bf16)b3.y;
        bfrag[14] = (__bf16)b3.z; bfrag[15] = (__bf16)b3.w;

        #pragma unroll
        for (int mt = 0; mt < 2; ++mt) {
            v8f acc = {};
            acc = __builtin_amdgcn_wmma_f32_16x16x32_bf16(
                false, afrag[mt], false, bfrag, (short)0, acc, false, false);
            const int row0 = mt * 16 + half * 8;   // D layout: VGPR r -> M = r + 8*half
            #pragma unroll
            for (int r = 0; r < 8; ++r)
                u_lds[(row0 + r) * STRIDE + col] = acc[r];
        }
    }

    __syncthreads();

    // ---- dynamic routing: wave w owns batch b = w; lane plays role n AND role d ----
    const int b = w;
    float* crow = c_lds + b * NUM_CAPS;
    float* orow = o_lds + b * NUM_CAPS;

    float row[DIM];   // role n=lane: u_hat[b, lane, d], d=0..31
    const float* rp = &u_lds[b * STRIDE + lane * DIM];
    #pragma unroll
    for (int q = 0; q < DIM / 4; ++q) {
        const float4 t = *reinterpret_cast<const float4*>(rp + 4 * q);
        row[4 * q + 0] = t.x; row[4 * q + 1] = t.y;
        row[4 * q + 2] = t.z; row[4 * q + 3] = t.w;
    }
    float col[NUM_CAPS];   // role d=lane: u_hat[b, n, lane], n=0..31 (conflict-free)
    #pragma unroll
    for (int n = 0; n < NUM_CAPS; ++n)
        col[n] = u_lds[b * STRIDE + n * DIM + lane];

    float blog = 0.0f;     // routing logit b[b, i, n=lane]
    float outv = 0.0f;     // out[b, i, d=lane]
    #pragma unroll
    for (int it = 0; it < ROUTINGS; ++it) {
        // softmax over n (two scalar 5-stage butterflies)
        float m = blog;
        #pragma unroll
        for (int off = 16; off; off >>= 1) m = fmaxf(m, __shfl_xor(m, off, 32));
        const float e = __expf(blog - m);
        float s = e;
        #pragma unroll
        for (int off = 16; off; off >>= 1) s += __shfl_xor(s, off, 32);
        crow[lane] = e / s;                       // publish c[b, n=lane]

        // out[d=lane] = relu(sum_n c[n] * col[n]) : broadcast b128 loads + FMAs
        outv = 0.0f;
        #pragma unroll
        for (int q = 0; q < NUM_CAPS / 4; ++q) {
            const float4 cv = *reinterpret_cast<const float4*>(crow + 4 * q);
            outv = fmaf(cv.x, col[4 * q + 0], outv);
            outv = fmaf(cv.y, col[4 * q + 1], outv);
            outv = fmaf(cv.z, col[4 * q + 2], outv);
            outv = fmaf(cv.w, col[4 * q + 3], outv);
        }
        outv = fmaxf(outv, 0.0f);

        if (it < ROUTINGS - 1) {
            orow[lane] = outv;                    // publish out[b, d=lane]
            float dot = 0.0f;                     // blog[n] += sum_d out[d]*row[d]
            #pragma unroll
            for (int q = 0; q < DIM / 4; ++q) {
                const float4 ov = *reinterpret_cast<const float4*>(orow + 4 * q);
                dot = fmaf(ov.x, row[4 * q + 0], dot);
                dot = fmaf(ov.y, row[4 * q + 1], dot);
                dot = fmaf(ov.z, row[4 * q + 2], dot);
                dot = fmaf(ov.w, row[4 * q + 3], dot);
            }
            blog += dot;
        }
    }

    // lane d holds out[b, i, d] directly
    out[((size_t)b * IN_CAPS + i) * DIM + lane] = outv;
}

extern "C" void kernel_launch(void* const* d_in, const int* in_sizes, int n_in,
                              void* d_out, int out_size, void* d_ws, size_t ws_size,
                              hipStream_t stream)
{
    const float* x = (const float*)d_in[0];
    const float* W = (const float*)d_in[1];
    float*       o = (float*)d_out;
    const size_t shmem =
        ((size_t)BATCH * STRIDE + 2 * BATCH * NUM_CAPS) * sizeof(float);  // ~136.5 KB
    (void)hipFuncSetAttribute((const void*)capsule_fused_kernel,
                              hipFuncAttributeMaxDynamicSharedMemorySize, (int)shmem);
    capsule_fused_kernel<<<IN_CAPS, 1024, shmem, stream>>>(x, W, o);
}
